// EDANet_81192061764341
// MI455X (gfx1250) — matrix operand order repaired and checked by reading, not compiled, not run on hardware
//
#include <hip/hip_runtime.h>
#include <math.h>

typedef __attribute__((ext_vector_type(16))) _Float16 v16h;
typedef __attribute__((ext_vector_type(8)))  float    v8f;
typedef __attribute__((ext_vector_type(4)))  unsigned int u32x4;
typedef __attribute__((address_space(1))) const u32x4* gu4p;

#define ROWS    32
#define THREADS 256
#define NBLOCKS 2048   // 65536 / 32

// ---- f16 weight workspace layout (element offsets), fragment-major order ----
#define OFF_QK   0        // K=256 N=32   xb -> logits (qw . pk^T folded, /sqrt(A))
#define OFF_FPW  8192     // K=128 N=256  head fusion
#define OFF_SE1  40960    // K=256 N=64
#define OFF_SE2  57344    // K=64  N=256
#define OFF_B1   73728    // K=256 N=256
#define OFF_W2   139264   // K=256 N=256  [b2_w | r2_w]
#define OFF_W3   204800   // K=128 N=128  [b3_w | r3_w]
#define OFF_C1   221184   // K=64  N=32
#define OFF_PVOW 223232   // (H*P, A) = (32,32)  pv . ow folded
#define OFF_QKB  224256   // 32  qb . pk^T /sqrt(A)
#define WS_TOTAL 224288

// Fragment = 512 f16 per-lane B operand, in this code's contiguous k order (as loadA):
//   j = ((kt*(N/16)+nt) << 9) + lane*16 + e  ->  k = kt*32 + (lane>>4)*16 + e
//                                               n = nt*16 + (lane&15)
__device__ __forceinline__ void frag_kn(int j, int N, int& k, int& n) {
  int frag = j >> 9, w = j & 511;
  int NT = N >> 4;
  int nt = frag % NT, kt = frag / NT;
  int l = w >> 4, e = w & 15;
  k = kt * 32 + ((l >> 4) << 4) + e;
  n = nt * 16 + (l & 15);
}

// ============ weight repack + algebraic pre-folding: f32 -> f16 fragments ============
__global__ void repack_weights(const float* __restrict__ qw,   // (4,256,32)
                               const float* __restrict__ qb,   // (4,32)
                               const float* __restrict__ pk,   // (4,8,32)
                               const float* __restrict__ pv,   // (4,8,32)
                               const float* __restrict__ ow,   // (4,32,32)
                               const float* __restrict__ fp_w, // (128,256)
                               const float* __restrict__ se_w1,// (256,64)
                               const float* __restrict__ se_w2,// (64,256)
                               const float* __restrict__ b1_w, // (256,256)
                               const float* __restrict__ b2_w, // (256,128)
                               const float* __restrict__ r2_w, // (256,128)
                               const float* __restrict__ b3_w, // (128,64)
                               const float* __restrict__ r3_w, // (128,64)
                               const float* __restrict__ c1_w, // (64,32)
                               _Float16* __restrict__ ws) {
  int i = blockIdx.x * blockDim.x + threadIdx.x;
  if (i >= WS_TOTAL) return;
  int k, n;
  float v;
  if (i < OFF_FPW) {                                   // QK = qw . pk^T / sqrt(A)
    frag_kn(i - OFF_QK, 32, k, n);
    int h = n >> 3, p = n & 7;
    float s = 0.f;
    for (int a = 0; a < 32; ++a) s += qw[h * 8192 + k * 32 + a] * pk[(h * 8 + p) * 32 + a];
    v = s * 0.17677669529663687f;
  } else if (i < OFF_SE1) {
    frag_kn(i - OFF_FPW, 256, k, n); v = fp_w[k * 256 + n];
  } else if (i < OFF_SE2) {
    frag_kn(i - OFF_SE1, 64, k, n);  v = se_w1[k * 64 + n];
  } else if (i < OFF_B1) {
    frag_kn(i - OFF_SE2, 256, k, n); v = se_w2[k * 256 + n];
  } else if (i < OFF_W2) {
    frag_kn(i - OFF_B1, 256, k, n);  v = b1_w[k * 256 + n];
  } else if (i < OFF_W3) {
    frag_kn(i - OFF_W2, 256, k, n);
    v = (n < 128) ? b2_w[k * 128 + n] : r2_w[k * 128 + (n - 128)];
  } else if (i < OFF_C1) {
    frag_kn(i - OFF_W3, 128, k, n);
    v = (n < 64) ? b3_w[k * 64 + n] : r3_w[k * 64 + (n - 64)];
  } else if (i < OFF_PVOW) {
    frag_kn(i - OFF_C1, 32, k, n);   v = c1_w[k * 32 + n];
  } else if (i < OFF_QKB) {                            // PVOW[hp][e] = pv . ow
    int j = i - OFF_PVOW, hp = j >> 5, e = j & 31, h = hp >> 3;
    float s = 0.f;
    for (int a = 0; a < 32; ++a) s += pv[hp * 32 + a] * ow[h * 1024 + a * 32 + e];
    v = s;
  } else {                                             // QKB = qb . pk^T / sqrt(A)
    int j = i - OFF_QKB, h = j >> 3;
    float s = 0.f;
    for (int a = 0; a < 32; ++a) s += qb[h * 32 + a] * pk[j * 32 + a];
    v = s * 0.17677669529663687f;
  }
  ws[i] = (_Float16)v;
}

// ===================== WMMA fragment helpers (16x16x32 f16) =====================
__device__ __forceinline__ v16h loadA(const _Float16* S, int m0, int k0, int lda) {
  int lane = threadIdx.x & 31;
  int m  = m0 + (lane & 15);
  int kh = (lane >> 4) * 16;
  const _Float16* p = S + m * lda + k0;
  v16h a;
#pragma unroll
  for (int v = 0; v < 8; ++v) {
    int kb = kh + v * 2;
    a[2 * v]     = p[kb];
    a[2 * v + 1] = p[kb + 1];
  }
  return a;
}
// B fragment pre-swizzled in ws: lane's 16 f16 contiguous; force AS(1) -> global_load_b128
__device__ __forceinline__ v16h loadBg(const _Float16* __restrict__ frag) {
  gu4p gp = (gu4p)(unsigned long long)(frag + (threadIdx.x & 31) * 16);
  union { v16h v; u32x4 q[2]; } u;
  u.q[0] = gp[0];
  u.q[1] = gp[1];
  return u.v;
}
__device__ __forceinline__ void storeD(_Float16* D, int m0, int n0, int ldd, v8f c) {
  int lane = threadIdx.x & 31;
  int n  = n0 + (lane & 15);
  int mh = (lane >> 4) * 8;
#pragma unroll
  for (int i = 0; i < 8; ++i)
    D[(m0 + mh + i) * ldd + n] = (_Float16)c[i];
}

// ========== tiled GEMM: sD[32][N] = sA[32][K] * Wg(frag-major), SW-pipelined =========
template <int K, int N>
__device__ void gemm_f16(const _Float16* __restrict__ Wg, const _Float16* sA, int lda,
                         _Float16* sD, int ldd) {
  const int wave = (int)(threadIdx.x >> 5);
  constexpr int NT  = N / 16, KT = K / 32;
  constexpr int TPW = (2 * NT + 7) / 8;
  constexpr bool FULL = ((2 * NT) % 8) == 0;   // every wave owns TPW tiles
  const int mt  = wave & 1;
  const int nt0 = wave >> 1;
  const bool active = FULL || (nt0 < NT);      // wave-uniform
  v8f acc[TPW];
#pragma unroll
  for (int i = 0; i < TPW; ++i) acc[i] = (v8f){0.f,0.f,0.f,0.f,0.f,0.f,0.f,0.f};

  v16h a_cur, a_nxt;
  v16h b_cur[TPW], b_nxt[TPW];
  if (active) {
    a_cur = loadA(sA, mt * 16, 0, lda);
#pragma unroll
    for (int i = 0; i < TPW; ++i) {
      int nt = nt0 + 4 * i;
      if (FULL || nt < NT) b_cur[i] = loadBg(Wg + (nt << 9));
    }
  }
#pragma unroll
  for (int kt = 0; kt < KT; ++kt) {
    if (active) {
      if (kt + 1 < KT) {                     // prefetch next K-step into registers
        a_nxt = loadA(sA, mt * 16, (kt + 1) * 32, lda);
#pragma unroll
        for (int i = 0; i < TPW; ++i) {
          int nt = nt0 + 4 * i;
          if (FULL || nt < NT) {
            const _Float16* frag = Wg + ((((kt + 1) * NT) + nt) << 9);
            if (kt + 2 < KT) __builtin_prefetch(frag + (NT << 9), 0, 1);
            b_nxt[i] = loadBg(frag);
          }
        }
      }
#pragma unroll
      for (int i = 0; i < TPW; ++i) {
        int nt = nt0 + 4 * i;
        if (FULL || nt < NT)
          acc[i] = __builtin_amdgcn_wmma_f32_16x16x32_f16(
              false, a_cur, false, b_cur[i], (short)0, acc[i], false, false);
      }
      if (kt + 1 < KT) {
        a_cur = a_nxt;
#pragma unroll
        for (int i = 0; i < TPW; ++i) {
          int nt = nt0 + 4 * i;
          if (FULL || nt < NT) b_cur[i] = b_nxt[i];
        }
      }
    }
  }
  if (active) {
#pragma unroll
    for (int i = 0; i < TPW; ++i) {
      int nt = nt0 + 4 * i;
      if (FULL || nt < NT) storeD(sD, mt * 16, nt * 16, ldd, acc[i]);
    }
  }
  __syncthreads();
}

// ===================== params =====================
struct Params {
  const float *x, *bn0_g, *bn0_b, *bn0_m, *bn0_v;
  const float *tw1, *tb1, *tw2, *tb2, *ob, *lng, *lnb;
  const float *fp_b, *fln_g, *fln_b, *se_b1, *se_b2;
  const float *b1_b, *b1_g, *b1_bb, *b1_m, *b1_v;
  const float *b2_b, *b2_g, *b2_bb, *b2_m, *b2_v, *r2_b;
  const float *b3_b, *b3_g, *b3_bb, *b3_m, *b3_v, *r3_b;
  const float *c1_b, *c2_w, *c2_b;
  const _Float16 *wQK, *wFP, *wSE1, *wSE2, *wB1, *wB2, *wB3, *wC1, *pvow, *qkb;
  float* out;
};

__device__ __forceinline__ float gelu_f(float x) {
  return 0.5f * x * (1.f + erff(x * 0.70710678118654752f));
}
__device__ __forceinline__ float sigmoid_f(float x) {
  return 1.f / (1.f + expf(-x));
}

// ===================== fused network kernel =====================
__global__ __launch_bounds__(THREADS) void edanet_fused(Params P) {
  __shared__ _Float16 sXb[ROWS * 256];   // bn'd input (kept for fusion residual)
  __shared__ _Float16 sAct[ROWS * 256];  // current activation
  __shared__ _Float16 sTmp[ROWS * 256];  // GEMM output scratch

  const int tid = threadIdx.x;
  const int r0  = blockIdx.x * ROWS;

  // ---- input BatchNorm -> f16 ----
  {
    const float4* x4 = (const float4*)(P.x + (size_t)r0 * 256);
    const float4* m4 = (const float4*)P.bn0_m;
    const float4* v4 = (const float4*)P.bn0_v;
    const float4* g4 = (const float4*)P.bn0_g;
    const float4* b4 = (const float4*)P.bn0_b;
    for (int e = tid; e < ROWS * 64; e += THREADS) {
      int d4 = e & 63;
      float4 xv = x4[e], m = m4[d4], vv = v4[d4], g = g4[d4], b = b4[d4];
      union { _Float16 h[4]; uint2 u; } pk4;
      pk4.h[0] = (_Float16)((xv.x - m.x) * rsqrtf(vv.x + 1e-5f) * g.x + b.x);
      pk4.h[1] = (_Float16)((xv.y - m.y) * rsqrtf(vv.y + 1e-5f) * g.y + b.y);
      pk4.h[2] = (_Float16)((xv.z - m.z) * rsqrtf(vv.z + 1e-5f) * g.z + b.z);
      pk4.h[3] = (_Float16)((xv.w - m.w) * rsqrtf(vv.w + 1e-5f) * g.w + b.w);
      ((uint2*)sXb)[e] = pk4.u;
    }
  }
  __syncthreads();

  // ---- logits = xb @ QK  (256 -> 32, qw/pk folded) ----
  gemm_f16<256, 32>(P.wQK, sXb, 256, sTmp, 256);

  // ---- per-head EDT attention: one thread per (row, head) ----
  if (tid < ROWS * 4) {
    const int row = tid >> 2, h = tid & 3;
    float lg[8];
    for (int p = 0; p < 8; ++p)
      lg[p] = (float)sTmp[row * 256 + h * 8 + p] + (float)P.qkb[h * 8 + p];
    // entropy from log-softmax
    float mx = -1e30f;
    for (int p = 0; p < 8; ++p) mx = fmaxf(mx, lg[p]);
    float ssum = 0.f;
    for (int p = 0; p < 8; ++p) ssum += expf(lg[p] - mx);
    float lse = logf(ssum) + mx;
    float ent = 0.f;
    for (int p = 0; p < 8; ++p) { float lp = lg[p] - lse; ent -= expf(lp) * lp; }
    ent *= 0.48089834696298783f;          // 1/ln(8)
    // temperature MLP
    float ta = 0.f;
    for (int t = 0; t < 32; ++t) {
      float hid = fmaxf(0.f, ent * P.tw1[h * 32 + t] + P.tb1[h * 32 + t]);
      ta += hid * P.tw2[h * 32 + t];
    }
    float tau = 0.1f + 4.9f * sigmoid_f(ta + P.tb2[h]);
    // tempered softmax
    float itau = 1.f / tau, m2 = -1e30f;
    for (int p = 0; p < 8; ++p) m2 = fmaxf(m2, lg[p] * itau);
    float at[8]; float asum = 0.f;
    for (int p = 0; p < 8; ++p) { at[p] = expf(lg[p] * itau - m2); asum += at[p]; }
    float inv_as = 1.f / asum;
    for (int p = 0; p < 8; ++p) at[p] *= inv_as;
    // comb = attn @ PVOW + ob ; per-head LayerNorm  (pv/ow folded)
    float o[32]; float mu = 0.f;
    for (int e = 0; e < 32; ++e) {
      float s = P.ob[h * 32 + e];
      for (int p = 0; p < 8; ++p) s += at[p] * (float)P.pvow[(h * 8 + p) * 32 + e];
      o[e] = s; mu += s;
    }
    mu *= (1.f / 32.f);
    float var = 0.f;
    for (int e = 0; e < 32; ++e) { float d = o[e] - mu; var += d * d; }
    float inv = rsqrtf(var * (1.f / 32.f) + 1e-5f);
    for (int e = 0; e < 32; ++e)
      sAct[row * 256 + h * 32 + e] =
          (_Float16)((o[e] - mu) * inv * P.lng[h * 32 + e] + P.lnb[h * 32 + e]);
  }
  __syncthreads();

  // ---- head fusion: comb @ fp_w + fp_b + xb, LayerNorm (8 threads/row + shfl) ----
  gemm_f16<128, 256>(P.wFP, sAct, 256, sTmp, 256);
  {
    int r = tid >> 3, sub = tid & 7;
    float vbuf[32];
    float s = 0.f, s2 = 0.f;
    for (int j = 0; j < 32; ++j) {
      int d = sub * 32 + j;
      float v = (float)sTmp[r * 256 + d] + P.fp_b[d] + (float)sXb[r * 256 + d];
      vbuf[j] = v; s += v; s2 += v * v;
    }
    for (int o = 1; o < 8; o <<= 1) { s += __shfl_xor(s, o, 32); s2 += __shfl_xor(s2, o, 32); }
    float mu = s * (1.f / 256.f);
    float inv = rsqrtf(s2 * (1.f / 256.f) - mu * mu + 1e-5f);
    for (int j = 0; j < 32; ++j) {
      int d = sub * 32 + j;
      sAct[r * 256 + d] = (_Float16)((vbuf[j] - mu) * inv * P.fln_g[d] + P.fln_b[d]);
    }
  }
  __syncthreads();

  // ---- SE gating ----  (xb is dead; reuse sXb as scratch)
  gemm_f16<256, 64>(P.wSE1, sAct, 256, sTmp, 256);
  for (int e = tid; e < ROWS * 64; e += THREADS) {
    int r = e >> 6, d = e & 63;
    sXb[r * 256 + d] = (_Float16)fmaxf(0.f, (float)sTmp[r * 256 + d] + P.se_b1[d]);
  }
  __syncthreads();
  gemm_f16<64, 256>(P.wSE2, sXb, 256, sTmp, 256);
  for (int e = tid; e < ROWS * 256; e += THREADS) {
    int d = e & 255;
    float se = sigmoid_f((float)sTmp[e] + P.se_b2[d]);
    sAct[e] = (_Float16)((float)sAct[e] * se);
  }
  __syncthreads();

  // ---- block 1: 256 -> 256, identity residual ----
  gemm_f16<256, 256>(P.wB1, sAct, 256, sTmp, 256);
  for (int e = tid; e < ROWS * 256; e += THREADS) {
    int d = e & 255;
    float v = (float)sTmp[e] + P.b1_b[d];
    v = (v - P.b1_m[d]) * rsqrtf(P.b1_v[d] + 1e-5f) * P.b1_g[d] + P.b1_bb[d];
    sAct[e] = (_Float16)(gelu_f(v) + (float)sAct[e]);
  }
  __syncthreads();

  // ---- block 2: 256 -> 128 (main || residual packed in one GEMM) ----
  gemm_f16<256, 256>(P.wB2, sAct, 256, sTmp, 256);
  for (int e = tid; e < ROWS * 128; e += THREADS) {
    int r = e >> 7, d = e & 127;
    float v = (float)sTmp[r * 256 + d] + P.b2_b[d];
    v = (v - P.b2_m[d]) * rsqrtf(P.b2_v[d] + 1e-5f) * P.b2_g[d] + P.b2_bb[d];
    float res = (float)sTmp[r * 256 + 128 + d] + P.r2_b[d];
    sAct[r * 256 + d] = (_Float16)(gelu_f(v) + res);
  }
  __syncthreads();

  // ---- block 3: 128 -> 64 (packed) ----
  gemm_f16<128, 128>(P.wB3, sAct, 256, sTmp, 256);
  for (int e = tid; e < ROWS * 64; e += THREADS) {
    int r = e >> 6, d = e & 63;
    float v = (float)sTmp[r * 256 + d] + P.b3_b[d];
    v = (v - P.b3_m[d]) * rsqrtf(P.b3_v[d] + 1e-5f) * P.b3_g[d] + P.b3_bb[d];
    float res = (float)sTmp[r * 256 + 64 + d] + P.r3_b[d];
    sAct[r * 256 + d] = (_Float16)(gelu_f(v) + res);
  }
  __syncthreads();

  // ---- classifier: 64 -> 32 (WMMA), then 32 -> 1 + sigmoid ----
  gemm_f16<64, 32>(P.wC1, sAct, 256, sTmp, 256);
  if (tid < ROWS) {
    int r = tid;
    float acc = 0.f;
    for (int j = 0; j < 32; ++j) {
      float t = (float)sTmp[r * 256 + j] + P.c1_b[j];
      acc += gelu_f(t) * P.c2_w[j];
    }
    P.out[r0 + r] = sigmoid_f(acc + P.c2_b[0]);
  }
}

// ===================== launch =====================
extern "C" void kernel_launch(void* const* d_in, const int* in_sizes, int n_in,
                              void* d_out, int out_size, void* d_ws, size_t ws_size,
                              hipStream_t stream) {
  (void)in_sizes; (void)n_in; (void)out_size; (void)ws_size;
  const float* F[51];
  for (int i = 0; i < 51; ++i) F[i] = (const float*)d_in[i];
  _Float16* ws = (_Float16*)d_ws;

  repack_weights<<<(WS_TOTAL + THREADS - 1) / THREADS, THREADS, 0, stream>>>(
      F[5], F[6], F[7], F[8], F[13], F[17], F[21], F[23], F[25],
      F[31], F[37], F[39], F[45], F[47], ws);

  Params P;
  P.x = F[0];  P.bn0_g = F[1]; P.bn0_b = F[2]; P.bn0_m = F[3]; P.bn0_v = F[4];
  P.tw1 = F[9]; P.tb1 = F[10]; P.tw2 = F[11]; P.tb2 = F[12];
  P.ob = F[14]; P.lng = F[15]; P.lnb = F[16];
  P.fp_b = F[18]; P.fln_g = F[19]; P.fln_b = F[20];
  P.se_b1 = F[22]; P.se_b2 = F[24];
  P.b1_b = F[26]; P.b1_g = F[27]; P.b1_bb = F[28]; P.b1_m = F[29]; P.b1_v = F[30];
  P.b2_b = F[32]; P.b2_g = F[33]; P.b2_bb = F[34]; P.b2_m = F[35]; P.b2_v = F[36];
  P.r2_b = F[38];
  P.b3_b = F[40]; P.b3_g = F[41]; P.b3_bb = F[42]; P.b3_m = F[43]; P.b3_v = F[44];
  P.r3_b = F[46];
  P.c1_b = F[48]; P.c2_w = F[49]; P.c2_b = F[50];
  P.wQK = ws + OFF_QK;   P.wFP = ws + OFF_FPW;
  P.wSE1 = ws + OFF_SE1; P.wSE2 = ws + OFF_SE2;
  P.wB1 = ws + OFF_B1;   P.wB2 = ws + OFF_W2;
  P.wB3 = ws + OFF_W3;   P.wC1 = ws + OFF_C1;
  P.pvow = ws + OFF_PVOW; P.qkb = ws + OFF_QKB;
  P.out = (float*)d_out;

  edanet_fused<<<NBLOCKS, THREADS, 0, stream>>>(P);
}
